// RGCN_15676630630845
// MI455X (gfx1250) — matrix operand order repaired and checked by reading, not compiled, not run on hardware
//
#include <hip/hip_runtime.h>

#define NNODES 100000
#define NEDGES 1600000
#define RREL   8

typedef __attribute__((ext_vector_type(16))) __bf16 v16bf;
typedef __attribute__((ext_vector_type(8)))  float  v8f;

// ---------------------------------------------------------------------------
// Zero workspace (float4-wide, grid-stride)
// ---------------------------------------------------------------------------
__global__ void zero_f4_kernel(float4* __restrict__ p, size_t n4) {
  size_t i = (size_t)blockIdx.x * blockDim.x + threadIdx.x;
  size_t stride = (size_t)gridDim.x * blockDim.x;
  float4 z = make_float4(0.f, 0.f, 0.f, 0.f);
  for (; i < n4; i += stride) p[i] = z;
}

// ---------------------------------------------------------------------------
// Per-edge scatter: summed[etype*N + dst, :] += x[src, :]; cnt[seg] += 1
// 16 threads per edge, float4 each -> coalesced 256B row gather.
// ---------------------------------------------------------------------------
__global__ void scatter_kernel(const float* __restrict__ x,
                               const int*  __restrict__ src,
                               const int*  __restrict__ dst,
                               const int*  __restrict__ et,
                               float* __restrict__ summed,
                               float* __restrict__ cnt,
                               int do_count) {
  long long tid = (long long)blockIdx.x * blockDim.x + threadIdx.x;
  int e = (int)(tid >> 4);
  if (e >= NEDGES) return;
  int lane = (int)(tid & 15);
  int s = src[e];
  int d = dst[e];
  int r = et[e];
  size_t seg = (size_t)r * NNODES + (size_t)d;
  const float4 v = ((const float4*)(x + (size_t)s * 64))[lane];
  float* b = summed + seg * 64 + (size_t)lane * 4;
  atomicAdd(b + 0, v.x);
  atomicAdd(b + 1, v.y);
  atomicAdd(b + 2, v.z);
  atomicAdd(b + 3, v.w);
  if (do_count && lane == 0) atomicAdd(cnt + seg, 1.0f);
}

// ---------------------------------------------------------------------------
// Pre-pack weights (R relation mats + root) into bf16 B-fragment layout.
// Fragment = one v_wmma B operand (32x16 bf16): lane n' in [0,16) holds
// column n = nt*16+n', K = kc*32 + (lane>>4)*16 .. +15 (16 contiguous K).
// Stored lane-major: packed[(frag*32 + lane)*16 + i].
// ---------------------------------------------------------------------------
__global__ void pack_weights_kernel(const float* __restrict__ W,      // [R,64,Nout]
                                    const float* __restrict__ Wroot,  // [64,Nout]
                                    __bf16* __restrict__ packed,
                                    int Nout, int NT) {
  int tid = blockIdx.x * blockDim.x + threadIdx.x;
  int nfrag = (RREL + 1) * 2 * NT;
  int frag = tid >> 5;
  if (frag >= nfrag) return;
  int lane = tid & 31;
  int nt = frag % NT;
  int kc = (frag / NT) & 1;
  int r  = frag / (2 * NT);
  int n  = nt * 16 + (lane & 15);
  int kstart = kc * 32 + (lane >> 4) * 16;
  const float* wsrc = (r < RREL) ? (W + (size_t)r * 64 * Nout) : Wroot;
  __bf16* dstp = packed + ((size_t)frag * 32 + lane) * 16;
#pragma unroll
  for (int i = 0; i < 16; ++i)
    dstp[i] = (__bf16)wsrc[(size_t)(kstart + i) * Nout + n];
}

// ---------------------------------------------------------------------------
// Fused RGCN dense stage:
//   out[n,:] = sum_r (summed[r,n,:]/max(cnt,1)) @ W_r  +  x[n,:] @ W_root + b
// K = (R+1)*64 = 576 -> 18 bf16 WMMA k-steps of 32. NT = Nout/16 column tiles.
// One wave = 16-row M tile; 8 waves / 256-thread block = 128 rows.
// ---------------------------------------------------------------------------
template <int NT>
__global__ __launch_bounds__(256)
void rgcn_gemm_kernel(const float* __restrict__ summed,
                      const float* __restrict__ cnt,
                      const float* __restrict__ xroot,
                      const __bf16* __restrict__ packedW,
                      const float* __restrict__ bias,
                      float* __restrict__ out,
                      int relu) {
  const int lane = threadIdx.x & 31;
  const int wave = threadIdx.x >> 5;
  const int rowbase = blockIdx.x * 128 + wave * 16;
  if (rowbase >= NNODES) return;   // wave-uniform: EXEC stays all-ones for WMMA
  const int half = lane >> 4;
  const int mrow = lane & 15;
  const int row = rowbase + mrow;

  v8f acc[NT];
#pragma unroll
  for (int t = 0; t < NT; ++t)
    acc[t] = (v8f){0.f, 0.f, 0.f, 0.f, 0.f, 0.f, 0.f, 0.f};

#pragma unroll
  for (int r = 0; r <= RREL; ++r) {
    const float* srcrow;
    float scale;
    if (r < RREL) {
      size_t seg = (size_t)r * NNODES + (size_t)row;
      srcrow = summed + seg * 64;
      scale = 1.0f / fmaxf(cnt[seg], 1.0f);
    } else {
      srcrow = xroot + (size_t)row * 64;
      scale = 1.0f;
    }
    if (r < RREL) {  // prefetch next relation's row (global_prefetch_b8)
      const float* nxt = (r + 1 < RREL)
                           ? summed + ((size_t)(r + 1) * NNODES + row) * 64
                           : xroot + (size_t)row * 64;
      __builtin_prefetch(nxt, 0, 3);
    }
#pragma unroll
    for (int kc = 0; kc < 2; ++kc) {
      // A fragment (16x32 bf16): lane half h -> K 16h..16h+15 contiguous (matches packed B)
      const int k0 = kc * 32 + half * 16;
      const float4 p0 = *(const float4*)(srcrow + k0);
      const float4 p1 = *(const float4*)(srcrow + k0 + 4);
      const float4 q0 = *(const float4*)(srcrow + k0 + 8);
      const float4 q1 = *(const float4*)(srcrow + k0 + 12);
      v16bf a;
      a[0]  = (__bf16)(p0.x * scale); a[1]  = (__bf16)(p0.y * scale);
      a[2]  = (__bf16)(p0.z * scale); a[3]  = (__bf16)(p0.w * scale);
      a[4]  = (__bf16)(p1.x * scale); a[5]  = (__bf16)(p1.y * scale);
      a[6]  = (__bf16)(p1.z * scale); a[7]  = (__bf16)(p1.w * scale);
      a[8]  = (__bf16)(q0.x * scale); a[9]  = (__bf16)(q0.y * scale);
      a[10] = (__bf16)(q0.z * scale); a[11] = (__bf16)(q0.w * scale);
      a[12] = (__bf16)(q1.x * scale); a[13] = (__bf16)(q1.y * scale);
      a[14] = (__bf16)(q1.z * scale); a[15] = (__bf16)(q1.w * scale);
#pragma unroll
      for (int nt = 0; nt < NT; ++nt) {
        const int frag = (r * 2 + kc) * NT + nt;
        v16bf b = *(const v16bf*)(packedW + ((size_t)frag * 32 + lane) * 16);
        acc[nt] = __builtin_amdgcn_wmma_f32_16x16x32_bf16(
            /*neg_a=*/false, a, /*neg_b=*/false, b,
            /*c_mod=*/(short)0, acc[nt], /*reuse_a=*/false, /*reuse_b=*/false);
      }
    }
  }

  // Epilogue: C layout -> VGPR j: lanes 0-15 M=j, lanes 16-31 M=j+8; N=lane%16
  const int Nout = NT * 16;
#pragma unroll
  for (int nt = 0; nt < NT; ++nt) {
    const int n = nt * 16 + mrow;
    const float bv = bias[n];
#pragma unroll
    for (int j = 0; j < 8; ++j) {
      int m = rowbase + (half ? j + 8 : j);
      float v = acc[nt][j] + bv;
      if (relu) v = fmaxf(v, 0.f);
      out[(size_t)m * Nout + n] = v;
    }
  }
}

// ---------------------------------------------------------------------------
// Launcher
// ---------------------------------------------------------------------------
extern "C" void kernel_launch(void* const* d_in, const int* in_sizes, int n_in,
                              void* d_out, int out_size, void* d_ws, size_t ws_size,
                              hipStream_t stream) {
  const float* X  = (const float*)d_in[0];
  const int*   A  = (const int*)d_in[1];
  const int*   et = (const int*)d_in[2];
  const float* W1 = (const float*)d_in[3];
  const float* r1 = (const float*)d_in[4];
  const float* b1 = (const float*)d_in[5];
  const float* W2 = (const float*)d_in[6];
  const float* r2 = (const float*)d_in[7];
  const float* b2 = (const float*)d_in[8];
  float* out = (float*)d_out;

  const int* srcI = A;            // A[0] = src
  const int* dstI = A + NEDGES;   // A[1] = dst

  // Workspace layout (all offsets 256B-aligned):
  //   summed [R*N*64 f32] | cnt [R*N f32] | h1 [N*64 f32] | pw1 bf16 | pw2 bf16
  char* ws = (char*)d_ws;
  const size_t summed_bytes = (size_t)RREL * NNODES * 64 * sizeof(float); // 204,800,000
  const size_t cnt_bytes    = (size_t)RREL * NNODES * sizeof(float);      //   3,200,000
  const size_t h1_bytes     = (size_t)NNODES * 64 * sizeof(float);        //  25,600,000
  float*  summed = (float*)ws;
  float*  cnt    = (float*)(ws + summed_bytes);
  float*  h1     = (float*)(ws + summed_bytes + cnt_bytes);
  __bf16* pw1    = (__bf16*)(ws + summed_bytes + cnt_bytes + h1_bytes);
  __bf16* pw2    = (__bf16*)(ws + summed_bytes + cnt_bytes + h1_bytes + 73728);

  // Pack weights into WMMA B-fragment layout (tiny, once per launch).
  pack_weights_kernel<<<9, 256, 0, stream>>>(W1, r1, pw1, 64, 4);  // 72 frags
  pack_weights_kernel<<<5, 256, 0, stream>>>(W2, r2, pw2, 32, 2);  // 36 frags

  const int scat_blocks = (int)(((long long)NEDGES * 16 + 255) / 256);
  const int gemm_blocks = (NNODES + 127) / 128;

  // ---- Layer 1 ----
  zero_f4_kernel<<<2048, 256, 0, stream>>>((float4*)summed,
                                           (summed_bytes + cnt_bytes) / 16);
  scatter_kernel<<<scat_blocks, 256, 0, stream>>>(X, srcI, dstI, et,
                                                  summed, cnt, /*count=*/1);
  rgcn_gemm_kernel<4><<<gemm_blocks, 256, 0, stream>>>(summed, cnt, X, pw1, b1,
                                                       h1, /*relu=*/1);

  // ---- Layer 2 (cnt depends only on graph -> reuse) ----
  zero_f4_kernel<<<2048, 256, 0, stream>>>((float4*)summed, summed_bytes / 16);
  scatter_kernel<<<scat_blocks, 256, 0, stream>>>(h1, srcI, dstI, et,
                                                  summed, cnt, /*count=*/0);
  rgcn_gemm_kernel<2><<<gemm_blocks, 256, 0, stream>>>(summed, cnt, h1, pw2, b2,
                                                       out, /*relu=*/0);
}
